// AttentionBlock_9680856285591
// MI455X (gfx1250) — matrix operand order repaired and checked by reading, not compiled, not run on hardware
//
#include <hip/hip_runtime.h>
#include <cstdint>

// ---------------------------------------------------------------------------
// Types for CDNA5 WMMA
// ---------------------------------------------------------------------------
typedef __attribute__((ext_vector_type(16))) _Float16 v16h;
typedef __attribute__((ext_vector_type(8)))  _Float16 v8h;
typedef __attribute__((ext_vector_type(8)))  float    v8f;
typedef int gv4i __attribute__((vector_size(16)));   // matches builtin param type

// Problem constants
#define BATCH   16
#define CH      512
#define SEQ     1024          // H*W = 32*32
#define HEADS   8
#define DK      64
#define DQKV    1536          // 3 * HEADS * DK
#define NGROUP  32
#define CPG     16            // CH / NGROUP

// ---------------------------------------------------------------------------
// CDNA5 async global->LDS copy (GLOBAL_LOAD_ASYNC_TO_LDS_B128, ASYNCcnt),
// with a synchronous fallback so the file always compiles.
// ---------------------------------------------------------------------------
#if defined(__HIP_DEVICE_COMPILE__) && \
    __has_builtin(__builtin_amdgcn_global_load_async_to_lds_b128) && \
    __has_builtin(__builtin_amdgcn_s_wait_asynccnt)
#define USE_ASYNC_LDS 1
#else
#define USE_ASYNC_LDS 0
#endif

static __device__ __forceinline__ void g2l_copy16(const _Float16* g, _Float16* l) {
#if USE_ASYNC_LDS
    // Param 0: addrspace(1) pointer to int4; param 1: addrspace(3) pointer.
    // Flat LDS address: low 32 bits are the LDS byte offset (aperture layout).
    __builtin_amdgcn_global_load_async_to_lds_b128(
        (__attribute__((address_space(1))) gv4i*)(uintptr_t)g,
        (__attribute__((address_space(3))) gv4i*)(uint32_t)(uintptr_t)l,
        /*offset=*/0, /*cpol=*/0);
#else
    *(v8h*)l = *(const v8h*)g;
#endif
}

template <int PENDING>
static __device__ __forceinline__ void async_wait() {
#if USE_ASYNC_LDS
    __builtin_amdgcn_s_wait_asynccnt(PENDING);
#endif
}

// ---------------------------------------------------------------------------
// WMMA helpers
// ---------------------------------------------------------------------------
static __device__ __forceinline__ v8f wmma_f16(v16h a, v16h b, v8f c) {
    return __builtin_amdgcn_wmma_f32_16x16x32_f16(
        false, a, false, b, (short)0, c, false, false);
}

// A-operand fragment: 16x32 f16 tile, rows m0..m0+15, K chunk k0..k0+31.
// lanes 0-15 row M=lane hold K {0..7,16..23}; lanes 16-31 row M=lane-16 hold
// K {8..15,24..31}.
static __device__ __forceinline__ v16h load_a_frag(const _Float16* base,
                                                   int m0, int ld, int k0) {
    int lane = threadIdx.x & 31;
    int hl   = lane >> 4;
    int r    = lane & 15;
    const _Float16* p = base + (size_t)(m0 + r) * ld + k0 + hl * 8;
    union { v16h v; v8h h[2]; } u;
    u.h[0] = *(const v8h*)(p);
    u.h[1] = *(const v8h*)(p + 16);
    return u.v;
}

// B-operand fragment: 32x16 f16 tile, column n contiguous (stored [N][K], ld).
// lanes 0-15 col N=lane K {0..7,16..23}; lanes 16-31 col N=lane-16 K {8..15,24..31}.
static __device__ __forceinline__ v16h load_b_frag(const _Float16* base,
                                                   int n0, int ld, int k0) {
    int lane = threadIdx.x & 31;
    int hl   = lane >> 4;
    int n    = n0 + (lane & 15);
    const _Float16* p = base + (size_t)n * ld + k0 + hl * 8;
    union { v16h v; v8h h[2]; } u;
    u.h[0] = *(const v8h*)(p);
    u.h[1] = *(const v8h*)(p + 16);
    return u.v;
}

static __device__ __forceinline__ float rowmax16(float v) {
    #pragma unroll
    for (int m = 1; m < 16; m <<= 1) v = fmaxf(v, __shfl_xor(v, m, 32));
    return v;
}
static __device__ __forceinline__ float rowsum16(float v) {
    #pragma unroll
    for (int m = 1; m < 16; m <<= 1) v += __shfl_xor(v, m, 32);
    return v;
}

// ---------------------------------------------------------------------------
// Kernel 0: convert fp32 weights -> f16 operand buffers
// ---------------------------------------------------------------------------
__global__ void k_convert_weights(const float* __restrict__ proj_w,
                                  const float* __restrict__ out_w,
                                  _Float16* __restrict__ wproj_h,
                                  _Float16* __restrict__ wout_h) {
    const int n1 = DQKV * CH;
    const int n2 = CH * CH;
    for (int i = blockIdx.x * blockDim.x + threadIdx.x; i < n1 + n2;
         i += gridDim.x * blockDim.x) {
        if (i < n1) wproj_h[i] = (_Float16)proj_w[i];
        else        wout_h[i - n1] = (_Float16)out_w[i - n1];
    }
}

// ---------------------------------------------------------------------------
// Kernel 1: GroupNorm -> f16 tokens [B, S, C]
// ---------------------------------------------------------------------------
__global__ void k_groupnorm(const float* __restrict__ x,
                            const float* __restrict__ gw,
                            const float* __restrict__ gb,
                            _Float16* __restrict__ tokens) {
    const int b  = blockIdx.x >> 5;
    const int g  = blockIdx.x & 31;
    const int c0 = g * CPG;
    const int tid = threadIdx.x;

    float sum = 0.f, sq = 0.f;
    for (int idx = tid; idx < CPG * SEQ; idx += 256) {
        int cl = idx >> 10;
        int s  = idx & (SEQ - 1);
        float v = x[(((size_t)b * CH + c0 + cl) << 10) + s];
        sum += v; sq += v * v;
    }
    __shared__ float r1[256], r2[256];
    r1[tid] = sum; r2[tid] = sq;
    __syncthreads();
    #pragma unroll
    for (int st = 128; st > 0; st >>= 1) {
        if (tid < st) { r1[tid] += r1[tid + st]; r2[tid] += r2[tid + st]; }
        __syncthreads();
    }
    const float inv_n = 1.0f / (float)(CPG * SEQ);
    float mean = r1[0] * inv_n;
    float var  = r2[0] * inv_n - mean * mean;
    float rstd = rsqrtf(var + 1e-5f);

    for (int idx = tid; idx < CPG * SEQ; idx += 256) {
        int cl = idx >> 10;
        int s  = idx & (SEQ - 1);
        int c  = c0 + cl;
        float v = x[(((size_t)b * CH + c) << 10) + s];
        float y = (v - mean) * rstd * gw[c] + gb[c];
        tokens[((size_t)b * SEQ + s) * CH + c] = (_Float16)y;
    }
}

// ---------------------------------------------------------------------------
// Kernel 2: QKV GEMM  [16384 x 512] x [512 x 1536]  (W stored [N][K])
// Block: 256 thr = 8 waves; tile 256M x 64N; wave = 32M x 64N.
// B tile (64x32 f16) double-buffered in LDS via async global->LDS.
// ---------------------------------------------------------------------------
__global__ void __launch_bounds__(256) k_qkv_gemm(
        const _Float16* __restrict__ tokens,
        const _Float16* __restrict__ wproj,
        const float* __restrict__ proj_b,
        _Float16* __restrict__ qh,
        _Float16* __restrict__ kh,
        _Float16* __restrict__ vt) {
    __shared__ __align__(16) _Float16 bstage[2][64][40];   // padded rows
    const int tid  = threadIdx.x;
    const int wave = tid >> 5;
    const int lane = tid & 31;
    const int hl   = lane >> 4;
    const int nc   = lane & 15;
    const int m0   = blockIdx.x * 256 + wave * 32;
    const int n0   = blockIdx.y * 64;

    // staging copy: 64 rows x 4 chunks of 16B
    const int cn = tid >> 2, cc = tid & 3;
    const _Float16* gsrc = wproj + (size_t)(n0 + cn) * CH + cc * 8;

    v8f acc[8] = {};
    g2l_copy16(gsrc, &bstage[0][cn][cc * 8]);

    for (int i = 0; i < CH / 32; ++i) {
        const int k0 = i * 32;
        if (i + 1 < CH / 32) {
            g2l_copy16(gsrc + (k0 + 32), &bstage[(i + 1) & 1][cn][cc * 8]);
            async_wait<1>();
        } else {
            async_wait<0>();
        }
        __syncthreads();

        if (k0 + 32 < CH)
            __builtin_prefetch(tokens + (size_t)(m0 + nc) * CH + k0 + 32, 0, 3);

        v16h bf[4];
        #pragma unroll
        for (int t = 0; t < 4; ++t)
            bf[t] = load_b_frag(&bstage[i & 1][0][0], t * 16, 40, 0);
        #pragma unroll
        for (int rb = 0; rb < 2; ++rb) {
            v16h a = load_a_frag(tokens, m0 + rb * 16, CH, k0);
            #pragma unroll
            for (int t = 0; t < 4; ++t)
                acc[rb * 4 + t] = wmma_f16(a, bf[t], acc[rb * 4 + t]);
        }
        __syncthreads();
    }

    #pragma unroll
    for (int rb = 0; rb < 2; ++rb) {
        #pragma unroll
        for (int t = 0; t < 4; ++t) {
            int n     = n0 + t * 16 + nc;
            int h     = n / 192;
            int rem   = n - h * 192;
            int which = rem >> 6;        // 0=q, 1=k, 2=v
            int d     = rem & 63;
            float bias = proj_b[n];
            #pragma unroll
            for (int r = 0; r < 8; ++r) {
                int m  = m0 + rb * 16 + r + 8 * hl;
                int bi = m >> 10;
                int s  = m & (SEQ - 1);
                _Float16 hv = (_Float16)(acc[rb * 4 + t][r] + bias);
                if (which == 0)
                    qh[((size_t)(bi * HEADS + h) * SEQ + s) * DK + d] = hv;
                else if (which == 1)
                    kh[((size_t)(bi * HEADS + h) * SEQ + s) * DK + d] = hv;
                else
                    vt[((size_t)(bi * HEADS + h) * DK + d) * SEQ + s] = hv;
            }
        }
    }
}

// ---------------------------------------------------------------------------
// Kernel 3: fused flash attention per (b,h).
// 8 waves x 16 query rows; 32-key tiles; K/V tiles double-buffered in LDS
// (shared by all 8 waves) via async copies; online softmax; P through LDS.
// ---------------------------------------------------------------------------
__global__ void __launch_bounds__(256) k_attention(
        const _Float16* __restrict__ q,
        const _Float16* __restrict__ k,
        const _Float16* __restrict__ vt,
        _Float16* __restrict__ attnout) {
    __shared__ __align__(16) _Float16 kstage[2][32][72];   // [key][dk], padded
    __shared__ __align__(16) _Float16 vstage[2][64][40];   // [d][key],  padded
    __shared__ __align__(16) _Float16 pbuf[8][16 * 40];    // per-wave P tile

    const int bh   = blockIdx.x;
    const int tid  = threadIdx.x;
    const int wave = tid >> 5;
    const int lane = tid & 31;
    const int hl   = lane >> 4;
    const int nc   = lane & 15;
    const int m0   = blockIdx.y * 128 + wave * 16;

    const _Float16* qb  = q  + (size_t)bh * SEQ * DK;
    const _Float16* kb  = k  + (size_t)bh * SEQ * DK;
    const _Float16* vtb = vt + (size_t)bh * DK * SEQ;
    _Float16* pw = &pbuf[wave][0];

    // staging copy indices: K tile 32 rows x 8 chunks; V tile 64 rows x 4 chunks
    const int kj = tid >> 3, kc = tid & 7;
    const int vd = tid >> 2, vc = tid & 3;
    const _Float16* kg = kb  + (size_t)kj * DK  + kc * 8;
    const _Float16* vg = vtb + (size_t)vd * SEQ + vc * 8;

    v16h aq0 = load_a_frag(qb, m0, DK, 0);
    v16h aq1 = load_a_frag(qb, m0, DK, 32);

    float mrow[8], lrow[8];
    #pragma unroll
    for (int r = 0; r < 8; ++r) { mrow[r] = -3.0e38f; lrow[r] = 0.f; }
    v8f o[4] = {};
    const float scale = 0.125f;   // DK^-0.5

    g2l_copy16(kg, &kstage[0][kj][kc * 8]);
    g2l_copy16(vg, &vstage[0][vd][vc * 8]);

    for (int i = 0; i < SEQ / 32; ++i) {
        const int j0  = i * 32;
        const int buf = i & 1;
        if (i + 1 < SEQ / 32) {
            g2l_copy16(kg + (size_t)(j0 + 32) * DK, &kstage[buf ^ 1][kj][kc * 8]);
            g2l_copy16(vg + (j0 + 32),              &vstage[buf ^ 1][vd][vc * 8]);
            async_wait<2>();
        } else {
            async_wait<0>();
        }
        __syncthreads();

        // ---- scores: 16 query rows x 32 keys
        const _Float16* ks = &kstage[buf][0][0];
        v8f s0 = {}, s1 = {};
        s0 = wmma_f16(aq0, load_b_frag(ks, 0,  72, 0),  s0);
        s0 = wmma_f16(aq1, load_b_frag(ks, 0,  72, 32), s0);
        s1 = wmma_f16(aq0, load_b_frag(ks, 16, 72, 0),  s1);
        s1 = wmma_f16(aq1, load_b_frag(ks, 16, 72, 32), s1);

        // ---- online softmax (row = r + 8*hl, cols across 16 lanes)
        float p0[8], p1[8], corr[8];
        #pragma unroll
        for (int r = 0; r < 8; ++r) {
            float a = s0[r] * scale;
            float b = s1[r] * scale;
            s0[r] = a; s1[r] = b;
            float rm = rowmax16(fmaxf(a, b));
            float nm = fmaxf(mrow[r], rm);
            corr[r] = __expf(mrow[r] - nm);
            mrow[r] = nm;
        }
        #pragma unroll
        for (int r = 0; r < 8; ++r) {
            p0[r] = __expf(s0[r] - mrow[r]);
            p1[r] = __expf(s1[r] - mrow[r]);
            float rs = rowsum16(p0[r] + p1[r]);
            lrow[r] = lrow[r] * corr[r] + rs;
        }
        #pragma unroll
        for (int t = 0; t < 4; ++t)
            #pragma unroll
            for (int r = 0; r < 8; ++r) o[t][r] *= corr[r];

        // ---- P: C-layout -> A-layout through per-wave LDS tile
        #pragma unroll
        for (int r = 0; r < 8; ++r) {
            int row = r + 8 * hl;
            pw[row * 40 + nc]      = (_Float16)p0[r];
            pw[row * 40 + 16 + nc] = (_Float16)p1[r];
        }
        __syncthreads();
        v16h ap = load_a_frag(pw, 0, 40, 0);

        // ---- O += P * V
        const _Float16* vs = &vstage[buf][0][0];
        #pragma unroll
        for (int t = 0; t < 4; ++t)
            o[t] = wmma_f16(ap, load_b_frag(vs, t * 16, 40, 0), o[t]);
        __syncthreads();   // reads done before next overwrite / async issue
    }

    const int bi = bh >> 3;
    const int h  = bh & 7;
    #pragma unroll
    for (int t = 0; t < 4; ++t) {
        #pragma unroll
        for (int r = 0; r < 8; ++r) {
            int srow = m0 + r + 8 * hl;
            int col  = h * DK + t * 16 + nc;
            float val = o[t][r] / lrow[r];
            attnout[((size_t)bi * SEQ + srow) * CH + col] = (_Float16)val;
        }
    }
}

// ---------------------------------------------------------------------------
// Kernel 4: out projection [16384 x 512] x [512 x 512] + bias + residual
// Same LDS-staged structure as kernel 2; fp32 NCHW output.
// ---------------------------------------------------------------------------
__global__ void __launch_bounds__(256) k_out_proj(
        const _Float16* __restrict__ attnout,
        const _Float16* __restrict__ wout,
        const float* __restrict__ out_b,
        const float* __restrict__ x,
        float* __restrict__ out) {
    __shared__ __align__(16) _Float16 bstage[2][64][40];
    const int tid  = threadIdx.x;
    const int wave = tid >> 5;
    const int lane = tid & 31;
    const int hl   = lane >> 4;
    const int nc   = lane & 15;
    const int m0   = blockIdx.x * 256 + wave * 32;
    const int n0   = blockIdx.y * 64;

    const int cn = tid >> 2, cc = tid & 3;
    const _Float16* gsrc = wout + (size_t)(n0 + cn) * CH + cc * 8;

    v8f acc[8] = {};
    g2l_copy16(gsrc, &bstage[0][cn][cc * 8]);

    for (int i = 0; i < CH / 32; ++i) {
        const int k0 = i * 32;
        if (i + 1 < CH / 32) {
            g2l_copy16(gsrc + (k0 + 32), &bstage[(i + 1) & 1][cn][cc * 8]);
            async_wait<1>();
        } else {
            async_wait<0>();
        }
        __syncthreads();

        if (k0 + 32 < CH)
            __builtin_prefetch(attnout + (size_t)(m0 + nc) * CH + k0 + 32, 0, 3);

        v16h bf[4];
        #pragma unroll
        for (int t = 0; t < 4; ++t)
            bf[t] = load_b_frag(&bstage[i & 1][0][0], t * 16, 40, 0);
        #pragma unroll
        for (int rb = 0; rb < 2; ++rb) {
            v16h a = load_a_frag(attnout, m0 + rb * 16, CH, k0);
            #pragma unroll
            for (int t = 0; t < 4; ++t)
                acc[rb * 4 + t] = wmma_f16(a, bf[t], acc[rb * 4 + t]);
        }
        __syncthreads();
    }

    #pragma unroll
    for (int rb = 0; rb < 2; ++rb) {
        #pragma unroll
        for (int t = 0; t < 4; ++t) {
            int n = n0 + t * 16 + nc;
            float bias = out_b[n];
            #pragma unroll
            for (int r = 0; r < 8; ++r) {
                int m  = m0 + rb * 16 + r + 8 * hl;
                int bi = m >> 10;
                int s  = m & (SEQ - 1);
                size_t oi = ((size_t)bi * CH + n) * SEQ + s;
                out[oi] = acc[rb * 4 + t][r] + bias + x[oi];
            }
        }
    }
}

// ---------------------------------------------------------------------------
// Host launcher
// ---------------------------------------------------------------------------
extern "C" void kernel_launch(void* const* d_in, const int* in_sizes, int n_in,
                              void* d_out, int out_size, void* d_ws, size_t ws_size,
                              hipStream_t stream) {
    const float* x      = (const float*)d_in[0];
    const float* norm_w = (const float*)d_in[1];
    const float* norm_b = (const float*)d_in[2];
    const float* proj_w = (const float*)d_in[3];
    const float* proj_b = (const float*)d_in[4];
    const float* out_w  = (const float*)d_in[5];
    const float* out_b  = (const float*)d_in[6];
    float* out = (float*)d_out;

    char* ws = (char*)d_ws;
    size_t off = 0;
    _Float16* wproj_h = (_Float16*)(ws + off); off += (size_t)DQKV * CH * 2;
    _Float16* wout_h  = (_Float16*)(ws + off); off += (size_t)CH * CH * 2;
    _Float16* tokens  = (_Float16*)(ws + off); off += (size_t)BATCH * SEQ * CH * 2;
    _Float16* qh      = (_Float16*)(ws + off); off += (size_t)BATCH * HEADS * SEQ * DK * 2;
    _Float16* kh      = (_Float16*)(ws + off); off += (size_t)BATCH * HEADS * SEQ * DK * 2;
    _Float16* vt      = (_Float16*)(ws + off); off += (size_t)BATCH * HEADS * DK * SEQ * 2;
    _Float16* attnout = (_Float16*)(ws + off); off += (size_t)BATCH * SEQ * CH * 2;

    k_convert_weights<<<1024, 256, 0, stream>>>(proj_w, out_w, wproj_h, wout_h);
    k_groupnorm<<<BATCH * NGROUP, 256, 0, stream>>>(x, norm_w, norm_b, tokens);
    k_qkv_gemm<<<dim3(BATCH * SEQ / 256, DQKV / 64), 256, 0, stream>>>(
        tokens, wproj_h, proj_b, qh, kh, vt);
    k_attention<<<dim3(BATCH * HEADS, SEQ / 128), 256, 0, stream>>>(
        qh, kh, vt, attnout);
    k_out_proj<<<dim3(BATCH * SEQ / 256, CH / 64), 256, 0, stream>>>(
        attnout, wout_h, out_b, x, out);
}
